// LRU_25348896981257
// MI455X (gfx1250) — compile-verified, matrix operand order checked by reading, not run on hardware
//
#include <hip/hip_runtime.h>

// ---------------------------------------------------------------------------
// LRU forward for MI455X (gfx1250): bf16 WMMA GEMMs + f32 chunked scan
//   1. k_prep        : lambda, lambda^128, bf16-packed weights
//   2. k_gemm_bu     : Bu = x @ [gB_re; gB_im]^T           (WMMA bf16)
//   3. k_carry       : per-chunk scan carries (read-only pass)
//   4. k_scan_combine: prefix over chunk carries -> S_c
//   5. k_scan_gemm   : fused: exact scan (f32) -> bf16 states in LDS ->
//                      y = [st interleaved | x] @ Wout^T   (WMMA bf16)
// ---------------------------------------------------------------------------

typedef __attribute__((ext_vector_type(16))) __bf16 v16bf;
typedef __attribute__((ext_vector_type(8)))  float  v8f;

constexpr int B_SZ  = 8;
constexpr int T     = 4096;
constexpr int D_IN  = 128;
constexpr int D_OUT = 128;
constexpr int NCH   = 256;                 // N state channels
constexpr int M_TOTAL = B_SZ * T;          // 32768 rows
constexpr int CHUNK  = 128;
constexpr int NCHUNK = T / CHUNK;          // 32
constexpr int KOUT   = 2 * NCH + D_IN;     // 640: [st interleaved | x]
constexpr int ROWSTRIDE = 520;             // LDS row stride in bf16 (1040 B):
                                           // 260 dwords -> 4-bank shift/row,
                                           // 16 A-frag rows hit distinct banks

// ---- workspace layout (bytes) ----
constexpr size_t SZ_BU    = (size_t)M_TOTAL * NCH * sizeof(float);      // 32 MB
constexpr size_t SZ_CARRY = (size_t)B_SZ * NCHUNK * NCH * sizeof(float2);
constexpr size_t OFF_BU_RE = 0;
constexpr size_t OFF_BU_IM = OFF_BU_RE + SZ_BU;
constexpr size_t OFF_CARRY = OFF_BU_IM + SZ_BU;
constexpr size_t OFF_SC    = OFF_CARRY + SZ_CARRY;
constexpr size_t OFF_LAM   = OFF_SC + SZ_CARRY;
constexpr size_t OFF_LAML  = OFF_LAM  + NCH * sizeof(float2);
constexpr size_t OFF_WB    = OFF_LAML + NCH * sizeof(float2);
constexpr size_t OFF_WOUT  = OFF_WB + (size_t)(2 * NCH) * D_IN * sizeof(__bf16);

// ---------------------------------------------------------------------------
// helpers
// ---------------------------------------------------------------------------
__device__ __forceinline__ unsigned bf16bits(float f) {
  union { float f; unsigned u; } v{f};
  return (v.u + 0x7FFFu + ((v.u >> 16) & 1u)) >> 16;   // round-nearest-even
}
__device__ __forceinline__ __bf16 f2bf(float f) {
  unsigned short h = (unsigned short)bf16bits(f);
  return __builtin_bit_cast(__bf16, h);
}

union BFrag { v16bf v; uint4 q[2]; };

// 16x32 fragment from an f32 row. Element j of lane l maps to
// K = (j>=8)*16 + ((l>>4)&1)*8 + (j&7); caller passes kb = ktile*32+(l>>4)*8.
__device__ __forceinline__ v16bf load_a_frag(const float* __restrict__ rowp, int kb) {
  const float4* p0 = (const float4*)(rowp + kb);
  const float4* p1 = (const float4*)(rowp + kb + 16);
  float4 f0 = p0[0], f1 = p0[1], f2 = p1[0], f3 = p1[1];
  float t[16] = {f0.x, f0.y, f0.z, f0.w, f1.x, f1.y, f1.z, f1.w,
                 f2.x, f2.y, f2.z, f2.w, f3.x, f3.y, f3.z, f3.w};
  v16bf a;
#pragma unroll
  for (int j = 0; j < 16; ++j) a[j] = f2bf(t[j]);
  return a;
}

__device__ __forceinline__ v16bf load_b_frag(const __bf16* __restrict__ rowp, int kb) {
  BFrag b;
  b.q[0] = *(const uint4*)(rowp + kb);
  b.q[1] = *(const uint4*)(rowp + kb + 16);
  return b.v;
}

// ---------------------------------------------------------------------------
// Kernel 1: parameter prep
//   Wb   [512][128]: [gamma*B_re ; gamma*B_im]
//   Wout [128][640]: k<512 -> (k&1 ? -C_im[d][k>>1] : C_re[d][k>>1]);
//                    k>=512 -> D[d][k-512]          (matches LDS state layout)
// ---------------------------------------------------------------------------
__global__ __launch_bounds__(256) void k_prep(
    const float* __restrict__ nu_log, const float* __restrict__ theta_log,
    const float* __restrict__ gamma_log,
    const float* __restrict__ B_re, const float* __restrict__ B_im,
    const float* __restrict__ C_re, const float* __restrict__ C_im,
    const float* __restrict__ Dm,
    float2* __restrict__ lam, float2* __restrict__ lamL,
    __bf16* __restrict__ Wb, __bf16* __restrict__ Wout)
{
  const int tid = threadIdx.x;
  for (int n = tid; n < NCH; n += blockDim.x) {
    float la = __expf(-__expf(nu_log[n]));
    float ph = __expf(theta_log[n]);
    float lr = la * __cosf(ph), li = la * __sinf(ph);
    lam[n] = make_float2(lr, li);
    float pr = lr, pi = li;                 // lambda^128 via 7 squarings
#pragma unroll
    for (int s = 0; s < 7; ++s) {
      float qr = pr * pr - pi * pi;
      float qi = 2.0f * pr * pi;
      pr = qr; pi = qi;
    }
    lamL[n] = make_float2(pr, pi);
  }
  for (int i = tid; i < 2 * NCH * D_IN; i += blockDim.x) {
    int r = i / D_IN, k = i % D_IN;
    int n = r & (NCH - 1);
    float g = __expf(gamma_log[n]);
    float v = (r < NCH) ? B_re[n * D_IN + k] : B_im[n * D_IN + k];
    Wb[i] = f2bf(g * v);
  }
  for (int i = tid; i < D_OUT * KOUT; i += blockDim.x) {
    int d = i / KOUT, k = i % KOUT;
    float v;
    if (k < 2 * NCH) {
      int n = k >> 1;
      v = (k & 1) ? -C_im[d * NCH + n] : C_re[d * NCH + n];
    } else {
      v = Dm[d * D_IN + (k - 2 * NCH)];
    }
    Wout[i] = f2bf(v);
  }
}

// ---------------------------------------------------------------------------
// Kernel 2: Bu = x @ Wb^T  -> bu_re [M][256], bu_im [M][256]
// One wave per 16-row strip; A fragments kept in regs, reused over 32 n-tiles.
// ---------------------------------------------------------------------------
__global__ __launch_bounds__(128) void k_gemm_bu(
    const float* __restrict__ x, const __bf16* __restrict__ Wb,
    float* __restrict__ bu_re, float* __restrict__ bu_im)
{
  const int lane = threadIdx.x & 31;
  const int wid  = blockIdx.x * (blockDim.x >> 5) + (threadIdx.x >> 5);
  const int m0   = wid * 16;
  const int hi   = lane >> 4;
  const int row  = m0 + (lane & 15);
  const float* xrow = x + (size_t)row * D_IN;

  v16bf a[4];
#pragma unroll
  for (int kt = 0; kt < 4; ++kt) a[kt] = load_a_frag(xrow, kt * 32 + hi * 8);

  for (int nt = 0; nt < 32; ++nt) {
    v8f acc = {};
    const int ncol = nt * 16 + (lane & 15);             // 0..511 = Wb row
    const __bf16* wrow = Wb + (size_t)ncol * D_IN;
#pragma unroll
    for (int kt = 0; kt < 4; ++kt) {
      v16bf b = load_b_frag(wrow, kt * 32 + hi * 8);
      acc = __builtin_amdgcn_wmma_f32_16x16x32_bf16(
          false, a[kt], false, b, (short)0, acc, false, false);
    }
    float* dst = (nt < 16) ? bu_re : bu_im;
    const int col = ncol & (NCH - 1);
#pragma unroll
    for (int r = 0; r < 8; ++r)
      dst[(size_t)(m0 + r + hi * 8) * NCH + col] = acc[r];
  }
}

// ---------------------------------------------------------------------------
// Kernel 3: per-chunk carries (read-only local scan, zero initial state).
// block = (b, chunk); thread = channel n -> coalesced per time step.
// ---------------------------------------------------------------------------
__global__ __launch_bounds__(256) void k_carry(
    const float2* __restrict__ lam,
    const float* __restrict__ bu_re, const float* __restrict__ bu_im,
    float2* __restrict__ carry)
{
  const int b = blockIdx.x / NCHUNK;
  const int c = blockIdx.x % NCHUNK;
  const int n = threadIdx.x;
  const float2 L = lam[n];
  float sr = 0.0f, si = 0.0f;
  size_t idx = ((size_t)(b * T + c * CHUNK)) * NCH + n;
  for (int i = 0; i < CHUNK; ++i, idx += NCH) {
    float ur = bu_re[idx], ui = bu_im[idx];
    float nr = L.x * sr - L.y * si + ur;
    float ni = L.x * si + L.y * sr + ui;
    sr = nr; si = ni;
  }
  carry[(size_t)(b * NCHUNK + c) * NCH + n] = make_float2(sr, si);
}

// ---------------------------------------------------------------------------
// Kernel 4: combine chunk carries -> S_c = true state entering chunk c.
// ---------------------------------------------------------------------------
__global__ __launch_bounds__(256) void k_scan_combine(
    const float2* __restrict__ lamL, const float2* __restrict__ carry,
    float2* __restrict__ Sc)
{
  const int b = blockIdx.x;
  const int n = threadIdx.x;
  const float2 LL = lamL[n];
  float sr = 0.0f, si = 0.0f;
  for (int c = 0; c < NCHUNK; ++c) {
    size_t i = (size_t)(b * NCHUNK + c) * NCH + n;
    Sc[i] = make_float2(sr, si);
    float2 cv = carry[i];
    float nr = LL.x * sr - LL.y * si + cv.x;
    float ni = LL.x * si + LL.y * sr + cv.y;
    sr = nr; si = ni;
  }
}

// ---------------------------------------------------------------------------
// Kernel 5: fused scan + output GEMM.
// Block = (b, chunk): 256 threads scan 256 channels for 128 steps with exact
// initial state S_c, writing bf16 states into LDS (k = 2n -> re, 2n+1 -> im,
// row stride 520 bf16). Then 8 waves x (16 rows x 128 cols x K=640) WMMA.
// y stored via LDS repack as coalesced b128.
// ---------------------------------------------------------------------------
__global__ __launch_bounds__(256) void k_scan_gemm(
    const float2* __restrict__ lam, const float2* __restrict__ Sc,
    const float* __restrict__ bu_re, const float* __restrict__ bu_im,
    const float* __restrict__ x, const __bf16* __restrict__ Wout,
    float* __restrict__ y)
{
  __shared__ __attribute__((aligned(16))) __bf16 lds[CHUNK * ROWSTRIDE]; // 130 KB
  const int b = blockIdx.x / NCHUNK;
  const int c = blockIdx.x % NCHUNK;
  const int row0 = b * T + c * CHUNK;          // global row of chunk start

  // ---- scan phase: one thread per channel, conflict-free b32 LDS stores ----
  {
    const int n = threadIdx.x;
    const float2 L = lam[n];
    float2 s = Sc[(size_t)(b * NCHUNK + c) * NCH + n];
    size_t idx = (size_t)row0 * NCH + n;
    unsigned* ldsw = (unsigned*)lds;
    for (int t = 0; t < CHUNK; ++t, idx += NCH) {
      float ur = bu_re[idx], ui = bu_im[idx];
      float nr = L.x * s.x - L.y * s.y + ur;
      float ni = L.x * s.y + L.y * s.x + ui;
      s.x = nr; s.y = ni;
      ldsw[t * (ROWSTRIDE / 2) + n] = bf16bits(nr) | (bf16bits(ni) << 16);
    }
  }
  __syncthreads();

  // ---- GEMM phase: wave w owns chunk rows [w*16, w*16+16) ----
  const int lane = threadIdx.x & 31;
  const int wv   = threadIdx.x >> 5;           // 0..7
  const int hi   = lane >> 4;
  const int t0   = wv * 16;
  const int mrow = t0 + (lane & 15);

  v8f acc[8] = {};
#pragma unroll
  for (int kt = 0; kt < KOUT / 32; ++kt) {     // 20 ktiles
    v16bf a;
    if (kt < 16) {                             // bf16 states from LDS
      const __bf16* ar = lds + (size_t)mrow * ROWSTRIDE + kt * 32 + hi * 8;
      BFrag f;
      f.q[0] = *(const uint4*)(ar);
      f.q[1] = *(const uint4*)(ar + 16);
      a = f.v;
    } else {                                   // x rows from global, f32->bf16
      a = load_a_frag(x + (size_t)(row0 + mrow) * D_IN, (kt - 16) * 32 + hi * 8);
    }
#pragma unroll
    for (int nt = 0; nt < 8; ++nt) {
      const __bf16* wrow = Wout + (size_t)(nt * 16 + (lane & 15)) * KOUT;
      v16bf bfr = load_b_frag(wrow, kt * 32 + hi * 8);
      acc[nt] = __builtin_amdgcn_wmma_f32_16x16x32_bf16(
          false, a, false, bfr, (short)0, acc[nt], false, false);
    }
  }

  // ---- repack through this wave's own (now dead) LDS rows, store b128 ----
  float* rep = (float*)(lds + (size_t)t0 * ROWSTRIDE);   // 16.6 KB >= 8 KB
#pragma unroll
  for (int nt = 0; nt < 8; ++nt)
#pragma unroll
    for (int r = 0; r < 8; ++r)
      rep[(r + hi * 8) * D_OUT + nt * 16 + (lane & 15)] = acc[nt][r];
  // DS ops are in-order within a wave; no barrier needed (wave-private rows).
  const int lrow = lane >> 1;                  // 0..15
  const int lcol = (lane & 1) * 64;            // 0 or 64
  const float4* src = (const float4*)(rep + lrow * D_OUT + lcol);
  float4* dst = (float4*)(y + (size_t)(row0 + t0 + lrow) * D_OUT + lcol);
#pragma unroll
  for (int q = 0; q < 16; ++q) dst[q] = src[q];
}

// ---------------------------------------------------------------------------
// host-side launch
// ---------------------------------------------------------------------------
extern "C" void kernel_launch(void* const* d_in, const int* in_sizes, int n_in,
                              void* d_out, int out_size, void* d_ws, size_t ws_size,
                              hipStream_t stream) {
  const float* x         = (const float*)d_in[0];
  const float* nu_log    = (const float*)d_in[1];
  const float* theta_log = (const float*)d_in[2];
  const float* gamma_log = (const float*)d_in[3];
  const float* B_re      = (const float*)d_in[4];
  const float* B_im      = (const float*)d_in[5];
  const float* C_re      = (const float*)d_in[6];
  const float* C_im      = (const float*)d_in[7];
  const float* Dm        = (const float*)d_in[8];
  float* y = (float*)d_out;

  char* ws = (char*)d_ws;
  float*  bu_re = (float*)(ws + OFF_BU_RE);
  float*  bu_im = (float*)(ws + OFF_BU_IM);
  float2* carry = (float2*)(ws + OFF_CARRY);
  float2* Sc    = (float2*)(ws + OFF_SC);
  float2* lam   = (float2*)(ws + OFF_LAM);
  float2* lamL  = (float2*)(ws + OFF_LAML);
  __bf16* Wb    = (__bf16*)(ws + OFF_WB);
  __bf16* Wout  = (__bf16*)(ws + OFF_WOUT);

  k_prep<<<1, 256, 0, stream>>>(nu_log, theta_log, gamma_log, B_re, B_im,
                                C_re, C_im, Dm, lam, lamL, Wb, Wout);

  k_gemm_bu<<<M_TOTAL / 16 / 4, 128, 0, stream>>>(x, Wb, bu_re, bu_im);

  k_carry<<<B_SZ * NCHUNK, NCH, 0, stream>>>(lam, bu_re, bu_im, carry);
  k_scan_combine<<<B_SZ, NCH, 0, stream>>>(lamL, carry, Sc);

  k_scan_gemm<<<B_SZ * NCHUNK, NCH, 0, stream>>>(lam, Sc, bu_re, bu_im,
                                                 x, Wout, y);
}
